// LlamaAttention_4896262717895
// MI455X (gfx1250) — compile-verified, matrix operand order checked by reading, not run on hardware
//
#include <hip/hip_runtime.h>
#include <hip/hip_bf16.h>
#include <stdint.h>

// Problem constants (BATCH=1)
static constexpr int kSEQ   = 2048;
static constexpr int kDIM   = 2048;
static constexpr int kHEADS = 16;
static constexpr int kHD    = 128;   // head dim

typedef __attribute__((ext_vector_type(16))) __bf16 v16bf;
typedef __attribute__((ext_vector_type(8)))  float  v8f;

// ---------------------------------------------------------------------------
// CDNA5 async global->LDS staging (ASYNCcnt-tracked). The builtin takes
// (v4i __device__* src, v4i __shared__* dst, imm offset, imm cpol) per the
// compiler diagnostic. Guarded so the file compiles even without it.
// ---------------------------------------------------------------------------
#if defined(__has_builtin)
#  if __has_builtin(__builtin_amdgcn_global_load_async_to_lds_b128)
#    define HAVE_ASYNC_LDS 1
#  endif
#endif
#ifndef HAVE_ASYNC_LDS
#  define HAVE_ASYNC_LDS 0
#endif

#if HAVE_ASYNC_LDS && defined(__has_builtin)
#  if __has_builtin(__builtin_amdgcn_s_wait_asynccnt)
#    define WAIT_ASYNC0() __builtin_amdgcn_s_wait_asynccnt(0)
#  else
#    define WAIT_ASYNC0() asm volatile("s_wait_asynccnt 0x0" ::: "memory")
#  endif
#else
#  define WAIT_ASYNC0() ((void)0)
#endif

typedef int v4i_t __attribute__((__vector_size__(16)));
typedef __attribute__((address_space(1))) v4i_t glb_v4i;
typedef __attribute__((address_space(3))) v4i_t lds_v4i;

// 16-byte global -> LDS stage (async when available)
__device__ __forceinline__ void stage16(const unsigned short* g, unsigned short* l) {
#if HAVE_ASYNC_LDS
  // generic -> AS1: full 64-bit VA; generic -> AS3: low 32 bits = LDS byte addr
  glb_v4i* gp = (glb_v4i*)(unsigned long long)g;
  lds_v4i* lp = (lds_v4i*)(unsigned int)(unsigned long long)l;
  __builtin_amdgcn_global_load_async_to_lds_b128(gp, lp, 0, 0);
#else
  *(uint4*)l = *(const uint4*)g;
#endif
}

__device__ __forceinline__ unsigned short f32_to_bf16(float f) {
  unsigned int u = __float_as_uint(f);
  u += 0x7FFFu + ((u >> 16) & 1u);      // round-to-nearest-even
  return (unsigned short)(u >> 16);
}

// ---------------------------------------------------------------------------
// A/B fragment load for V_WMMA_*_16x16x32 (16-bit), wave32.
// Lane L: row = L&15 ; lanes 0-15 hold K {0..7,16..23}, lanes 16-31 {8..15,24..31}.
// ---------------------------------------------------------------------------
__device__ __forceinline__ v16bf load_frag_g(const unsigned short* __restrict__ base,
                                             int row, int k0, int ld) {
  const int lane = threadIdx.x & 31;
  const int r    = lane & 15;
  const int koff = (lane >> 4) << 3;    // 0 or 8
  const unsigned short* p = base + (size_t)(row + r) * ld + (k0 + koff);
  union { uint4 u[2]; v16bf v; } t;
  t.u[0] = *(const uint4*)(p);
  t.u[1] = *(const uint4*)(p + 16);
  return t.v;
}

__device__ __forceinline__ v16bf load_frag_l(const unsigned short* base, int row0, int ld) {
  const int lane = threadIdx.x & 31;
  const int r    = lane & 15;
  const int koff = (lane >> 4) << 3;
  const unsigned short* p = base + (row0 + r) * ld + koff;
  union { uint4 u[2]; v16bf v; } t;
  t.u[0] = *(const uint4*)(p);
  t.u[1] = *(const uint4*)(p + 16);
  return t.v;
}

__device__ __forceinline__ v8f wmma_bf16(v16bf a, v16bf b, v8f c) {
  // (neg_a, A, neg_b, B, c_mod, C, reuse_a, reuse_b)
  return __builtin_amdgcn_wmma_f32_16x16x32_bf16(false, a, false, b, (short)0, c,
                                                 false, false);
}

// ---------------------------------------------------------------------------
// fp32 -> bf16 conversion (4 elements / thread)
// ---------------------------------------------------------------------------
__global__ __launch_bounds__(256) void cvt_f32_bf16_kernel(
    const float* __restrict__ src, unsigned short* __restrict__ dst) {
  int i = (blockIdx.x * 256 + threadIdx.x) * 4;
  float4 f = *(const float4*)(src + i);
  union { unsigned short s[4]; uint2 u; } o;
  o.s[0] = f32_to_bf16(f.x); o.s[1] = f32_to_bf16(f.y);
  o.s[2] = f32_to_bf16(f.z); o.s[3] = f32_to_bf16(f.w);
  *(uint2*)(dst + i) = o.u;
}

// ---------------------------------------------------------------------------
// C[M,N] = A[M,K] @ B[N,K]^T  (bf16 in, f32 accumulate)
// MODE 0: store bf16 [M,N]
// MODE 1: apply RoPE, store bf16 [M,N]          (Q, K projections)
// MODE 2: store bf16 transposed [N,M]           (V -> Vt for PV B-fragments)
// MODE 3: store f32  [M,N]                      (final output projection)
//
// Workgroup = 256 threads (8 waves) computes a 128x128 block. Per K-step (32)
// the WG stages A[128x32] + B[128x32] into LDS with async b128 copies
// (double-buffered, rows padded to 80B for conflict-free ds_load_b128), then
// each wave runs 8 WMMAs on its 32x64 sub-block from LDS.
// ---------------------------------------------------------------------------
static constexpr int kLROW = 40;  // LDS row pitch in bf16 elems (32 data + 8 pad)

template <int MODE>
__global__ __launch_bounds__(256) void gemm_bt_kernel(
    const unsigned short* __restrict__ A,
    const unsigned short* __restrict__ B,
    void* __restrict__ D,
    const float* __restrict__ sinT,
    const float* __restrict__ cosT) {
  __shared__ unsigned short slab[2][256 * kLROW];  // 2 x 20KB

  const int tid  = threadIdx.x;
  const int lane = tid & 31;
  const int wave = tid >> 5;
  const int m0 = blockIdx.y * 128;
  const int n0 = blockIdx.x * 128;
  const int wm = (wave >> 1) * 32;   // wave row offset within 128
  const int wn = (wave & 1) * 64;    // wave col offset within 128

  // Stage one K-slab: rows 0..127 = A(m0+row), rows 128..255 = B(n0+row-128).
  // 1024 16B chunks, 4 per thread.
  auto issue_stage = [&](int buf, int k0) {
#pragma unroll
    for (int c = 0; c < 4; c++) {
      int chunk = tid + c * 256;
      int row = chunk >> 2;
      int sub = (chunk & 3) * 8;  // element offset within row
      const unsigned short* src =
          (row < 128) ? (A + (size_t)(m0 + row) * kDIM + k0 + sub)
                      : (B + (size_t)(n0 + row - 128) * kDIM + k0 + sub);
      stage16(src, &slab[buf][row * kLROW + sub]);
    }
  };

  v8f acc[2][4];
#pragma unroll
  for (int i = 0; i < 2; i++)
#pragma unroll
    for (int j = 0; j < 4; j++) acc[i][j] = (v8f){0, 0, 0, 0, 0, 0, 0, 0};

  issue_stage(0, 0);
  int buf = 0;
  for (int k0 = 0; k0 < kDIM; k0 += 32) {
    WAIT_ASYNC0();        // my async copies into slab[buf] are done
    __syncthreads();      // everyone's copies are done -> slab[buf] valid
    if (k0 + 32 < kDIM) issue_stage(buf ^ 1, k0 + 32);  // overlap next fill

    const unsigned short* As = &slab[buf][0];
    const unsigned short* Bs = &slab[buf][128 * kLROW];
    v16bf a0 = load_frag_l(As, wm, kLROW);
    v16bf a1 = load_frag_l(As, wm + 16, kLROW);
#pragma unroll
    for (int t = 0; t < 4; t++) {
      v16bf bf = load_frag_l(Bs, wn + 16 * t, kLROW);
      acc[0][t] = wmma_bf16(a0, bf, acc[0][t]);
      acc[1][t] = wmma_bf16(a1, bf, acc[1][t]);
    }
    __syncthreads();      // all reads of slab[buf] done before it is refilled
    buf ^= 1;
  }

  // Epilogue. C layout: VGPR r -> row m = r (+8 for lanes 16..31); lane%16 -> col.
  const int nlane = lane & 15;
  const int hsel  = lane >> 4;
#pragma unroll
  for (int mt = 0; mt < 2; mt++)
#pragma unroll
    for (int t = 0; t < 4; t++)
#pragma unroll
      for (int r = 0; r < 8; r++) {
        float v = acc[mt][t][r];
        int row = m0 + wm + mt * 16 + r + 8 * hsel;
        int col = n0 + wn + t * 16 + nlane;
        if (MODE == 1) {
          // RoPE: channel pair (2i,2i+1) within first half of head dim.
          float other = __shfl_xor(v, 1, 32);  // rotation partner (adjacent lane)
          int ch = col & (kHD - 1);
          if (ch < kHD / 2) {
            int pair = ch >> 1;                      // < 32
            float s = sinT[row * (kHD / 2) + pair];  // sin stride = HD/2 = 64
            float c = cosT[row * (kHD / 2) + pair];
            bool even = (ch & 1) == 0;
            float x1 = even ? v : other;
            float x2 = even ? other : v;
            v = even ? (x1 * c - x2 * s) : (x1 * s + x2 * c);
          }
        }
        if (MODE == 0 || MODE == 1) {
          ((unsigned short*)D)[(size_t)row * kDIM + col] = f32_to_bf16(v);
        } else if (MODE == 2) {
          ((unsigned short*)D)[(size_t)col * kSEQ + row] = f32_to_bf16(v);
        } else {
          ((float*)D)[(size_t)row * kDIM + col] = v;
        }
      }
}

// ---------------------------------------------------------------------------
// Flash attention: one wave per (16-query tile, head). Causal, online softmax.
// Q,K: bf16 [S,D] (post-RoPE). Vt: bf16 [D,S]. Ctx out: bf16 [S,D].
// ---------------------------------------------------------------------------
__global__ __launch_bounds__(32) void flash_attn_kernel(
    const unsigned short* __restrict__ Q,
    const unsigned short* __restrict__ Kb,
    const unsigned short* __restrict__ Vt,
    unsigned short* __restrict__ Ctx) {
  __shared__ unsigned short Pld[16 * 32];  // P tile relayout: C-layout -> A-layout
  const int lane  = threadIdx.x & 31;
  const int nlane = lane & 15;
  const int hsel  = lane >> 4;
  const int q0 = blockIdx.x * 16;
  const int cb = blockIdx.y * kHD;
  const float scale = 0.088388347648318447f;  // 1/sqrt(128)
  const float NEG = -3.0e38f;

  v16bf qf[4];
#pragma unroll
  for (int i = 0; i < 4; i++) qf[i] = load_frag_g(Q, q0, cb + 32 * i, kDIM);

  v8f acc[8];
#pragma unroll
  for (int t = 0; t < 8; t++) acc[t] = (v8f){0, 0, 0, 0, 0, 0, 0, 0};
  float mrow[8], lrow[8];
#pragma unroll
  for (int r = 0; r < 8; r++) { mrow[r] = NEG; lrow[r] = 0.0f; }

  for (int kb = 0; kb < q0 + 16; kb += 32) {  // causal: keys 0 .. q0+15
    // ---- scores: 16q x 32keys, contraction over head dim (128) ----
    v8f sc[2];
    sc[0] = (v8f){0, 0, 0, 0, 0, 0, 0, 0};
    sc[1] = (v8f){0, 0, 0, 0, 0, 0, 0, 0};
#pragma unroll
    for (int t = 0; t < 2; t++)
#pragma unroll
      for (int i = 0; i < 4; i++) {
        v16bf kf = load_frag_g(Kb, kb + 16 * t, cb + 32 * i, kDIM);
        sc[t] = wmma_bf16(qf[i], kf, sc[t]);
      }
    // ---- scale + causal mask + row max ----
    float mnew[8];
#pragma unroll
    for (int r = 0; r < 8; r++) mnew[r] = mrow[r];
#pragma unroll
    for (int t = 0; t < 2; t++)
#pragma unroll
      for (int r = 0; r < 8; r++) {
        int key  = kb + t * 16 + nlane;
        int qrow = q0 + r + 8 * hsel;
        float sv = (key <= qrow) ? sc[t][r] * scale : NEG;
        sc[t][r] = sv;
        mnew[r] = fmaxf(mnew[r], sv);
      }
#pragma unroll
    for (int r = 0; r < 8; r++) {  // reduce over the 16-lane column group
      mnew[r] = fmaxf(mnew[r], __shfl_xor(mnew[r], 1, 32));
      mnew[r] = fmaxf(mnew[r], __shfl_xor(mnew[r], 2, 32));
      mnew[r] = fmaxf(mnew[r], __shfl_xor(mnew[r], 4, 32));
      mnew[r] = fmaxf(mnew[r], __shfl_xor(mnew[r], 8, 32));
    }
    // ---- rescale running state ----
    float rs[8];
#pragma unroll
    for (int r = 0; r < 8; r++) {
      float alpha = __expf(mrow[r] - mnew[r]);
      lrow[r] *= alpha;
#pragma unroll
      for (int t = 0; t < 8; t++) acc[t][r] *= alpha;
      mrow[r] = mnew[r];
      rs[r] = 0.0f;
    }
    // ---- P = exp(sc - m), row sums, stage to LDS (row-major 16x32) ----
#pragma unroll
    for (int t = 0; t < 2; t++)
#pragma unroll
      for (int r = 0; r < 8; r++) {
        float p = __expf(sc[t][r] - mnew[r]);
        rs[r] += p;
        Pld[(r + 8 * hsel) * 32 + t * 16 + nlane] = f32_to_bf16(p);
      }
#pragma unroll
    for (int r = 0; r < 8; r++) {
      rs[r] += __shfl_xor(rs[r], 1, 32);
      rs[r] += __shfl_xor(rs[r], 2, 32);
      rs[r] += __shfl_xor(rs[r], 4, 32);
      rs[r] += __shfl_xor(rs[r], 8, 32);
      lrow[r] += rs[r];
    }
    // ---- acc += P(16x32) @ V(32x128); LDS in-order per wave, no barrier ----
    v16bf pf = load_frag_l(Pld, 0, 32);
#pragma unroll
    for (int t = 0; t < 8; t++) {
      v16bf vf = load_frag_g(Vt, cb + t * 16, kb, kSEQ);
      acc[t] = wmma_bf16(pf, vf, acc[t]);
    }
  }

  // ---- normalize and store context ----
#pragma unroll
  for (int r = 0; r < 8; r++) {
    float inv = 1.0f / lrow[r];
    int row = q0 + r + 8 * hsel;
#pragma unroll
    for (int t = 0; t < 8; t++) {
      int col = cb + t * 16 + nlane;
      Ctx[(size_t)row * kDIM + col] = f32_to_bf16(acc[t][r] * inv);
    }
  }
}

// ---------------------------------------------------------------------------
// Host launcher. Workspace layout (bf16 buffers), total 72 MB:
//  [0,8)xb [8,16)Wq [16,24)Wk [24,32)Wv [32,40)Wo
//  [40,48)Q [48,56)K [56,64)Vt [64,72)Ctx   (MB offsets)
// ---------------------------------------------------------------------------
extern "C" void kernel_launch(void* const* d_in, const int* in_sizes, int n_in,
                              void* d_out, int out_size, void* d_ws, size_t ws_size,
                              hipStream_t stream) {
  (void)in_sizes; (void)n_in; (void)out_size; (void)ws_size;
  const float* x    = (const float*)d_in[0];
  const float* Wq   = (const float*)d_in[1];
  const float* Wk   = (const float*)d_in[2];
  const float* Wv   = (const float*)d_in[3];
  const float* Wo   = (const float*)d_in[4];
  const float* sinT = (const float*)d_in[5];
  const float* cosT = (const float*)d_in[6];
  float* out = (float*)d_out;

  char* ws = (char*)d_ws;
  const size_t MB = 1ull << 20;
  unsigned short* xb  = (unsigned short*)(ws + 0 * MB);
  unsigned short* Wqb = (unsigned short*)(ws + 8 * MB);
  unsigned short* Wkb = (unsigned short*)(ws + 16 * MB);
  unsigned short* Wvb = (unsigned short*)(ws + 24 * MB);
  unsigned short* Wob = (unsigned short*)(ws + 32 * MB);
  unsigned short* Qb  = (unsigned short*)(ws + 40 * MB);
  unsigned short* Kbf = (unsigned short*)(ws + 48 * MB);
  unsigned short* Vt  = (unsigned short*)(ws + 56 * MB);
  unsigned short* Ctx = (unsigned short*)(ws + 64 * MB);

  const int NE = kSEQ * kDIM;            // 4M elements
  dim3 cvtG(NE / (256 * 4)), cvtB(256);
  cvt_f32_bf16_kernel<<<cvtG, cvtB, 0, stream>>>(x,  xb);
  cvt_f32_bf16_kernel<<<cvtG, cvtB, 0, stream>>>(Wq, Wqb);
  cvt_f32_bf16_kernel<<<cvtG, cvtB, 0, stream>>>(Wk, Wkb);
  cvt_f32_bf16_kernel<<<cvtG, cvtB, 0, stream>>>(Wv, Wvb);
  cvt_f32_bf16_kernel<<<cvtG, cvtB, 0, stream>>>(Wo, Wob);

  dim3 gG(kDIM / 128, kSEQ / 128), gB(256);
  gemm_bt_kernel<1><<<gG, gB, 0, stream>>>(xb, Wqb, Qb,  sinT, cosT);       // Q + RoPE
  gemm_bt_kernel<1><<<gG, gB, 0, stream>>>(xb, Wkb, Kbf, sinT, cosT);       // K + RoPE
  gemm_bt_kernel<2><<<gG, gB, 0, stream>>>(xb, Wvb, Vt,  nullptr, nullptr); // V -> Vt

  flash_attn_kernel<<<dim3(kSEQ / 16, kHEADS), dim3(32), 0, stream>>>(Qb, Kbf, Vt, Ctx);

  gemm_bt_kernel<3><<<gG, gB, 0, stream>>>(Ctx, Wob, out, nullptr, nullptr); // O proj
}
